// DeepseekV2Attention_4252017623204
// MI455X (gfx1250) — matrix operand order repaired and checked by reading, not compiled, not run on hardware
//
#include <hip/hip_runtime.h>
#include <hip/hip_bf16.h>

typedef __attribute__((ext_vector_type(16))) __bf16 v16bf;
typedef __attribute__((ext_vector_type(8)))  __bf16 v8bf;
typedef __attribute__((ext_vector_type(8)))  float  v8f;

static __device__ inline v8f zero8() {
    v8f z = {0.f,0.f,0.f,0.f,0.f,0.f,0.f,0.f};
    return z;
}

// ---- CDNA5 async global->LDS copy (16B per lane, ASYNCcnt-tracked) ----
static __device__ inline void async_copy16(const void* lds_ptr, const void* gptr) {
    unsigned lds_off = (unsigned)(unsigned long long)(uintptr_t)lds_ptr;
    unsigned long long ga = (unsigned long long)(uintptr_t)gptr;
    asm volatile("global_load_async_to_lds_b128 %0, %1, off"
                 :: "v"(lds_off), "v"(ga)
                 : "memory");
}
static __device__ inline void wait_async0() {
    asm volatile("s_wait_asynccnt 0x0" ::: "memory");
}

// ---- WMMA fragment loaders (bf16, 16x16x32), contiguous k order (k = 16*(lane/16) + s) in A and B ----

// A operand: 16x32 (MxK). lane L: row = L&15 ; K = (L>>4)*16 .. +15
static __device__ inline v16bf load_frag_a(const __bf16* smem, int stride,
                                           int row0, int k0, int lane) {
    int r  = row0 + (lane & 15);
    int kb = k0 + ((lane >> 4) << 4);
    const v8bf* p0 = (const v8bf*)(smem + (long)r * stride + kb);
    const v8bf* p1 = (const v8bf*)(smem + (long)r * stride + kb + 8);
    v8bf lo = *p0, hi = *p1;
    v16bf a;
#pragma unroll
    for (int i = 0; i < 8; ++i) { a[i] = lo[i]; a[i + 8] = hi[i]; }
    return a;
}

// B operand: 32x16 (KxN), source stored N-major rows of K ([n][k]).
static __device__ inline v16bf load_frag_bn(const __bf16* smem, int stride,
                                            int n0, int k0, int lane) {
    int n  = n0 + (lane & 15);
    int kb = k0 + ((lane >> 4) << 4);
    const v8bf* p0 = (const v8bf*)(smem + (long)n * stride + kb);
    const v8bf* p1 = (const v8bf*)(smem + (long)n * stride + kb + 8);
    v8bf lo = *p0, hi = *p1;
    v16bf b;
#pragma unroll
    for (int i = 0; i < 8; ++i) { b[i] = lo[i]; b[i + 8] = hi[i]; }
    return b;
}

// B operand from K-major storage ([k][n]) — used for P @ V (V is key-major).
static __device__ inline v16bf load_frag_bk(const __bf16* smem, int stride,
                                            int n0, int k0, int lane) {
    int n  = n0 + (lane & 15);
    int kb = k0 + ((lane >> 4) << 4);
    v16bf b;
#pragma unroll
    for (int j = 0; j < 16; ++j) b[j] = smem[(long)(kb + j) * stride + n];
    return b;
}

static __device__ inline v8f wmma_bf16(v16bf a, v16bf b, v8f c) {
    return __builtin_amdgcn_wmma_f32_16x16x32_bf16(
        false, a, false, b, (short)0, c, false, false);
}

// =====================================================================
// f32 -> bf16 streaming convert (n multiple of 4)
// =====================================================================
__global__ __launch_bounds__(256) void k_f32_to_bf16(
    const float* __restrict__ in, __bf16* __restrict__ out, long n)
{
    long stride = (long)gridDim.x * 256 * 4;
    for (long i = ((long)blockIdx.x * 256 + threadIdx.x) * 4; i < n; i += stride) {
        float4 f = *(const float4*)(in + i);
        out[i + 0] = (__bf16)f.x;
        out[i + 1] = (__bf16)f.y;
        out[i + 2] = (__bf16)f.z;
        out[i + 3] = (__bf16)f.w;
    }
}

// =====================================================================
// GEMM: C[M,N](f32) = A[M,K](bf16) * W[N,K](bf16)^T
// 256 thr = 8 waves, 128x128 tile, K-step 32, async-LDS double buffer.
// =====================================================================
__global__ __launch_bounds__(256) void k_gemm_bf16(
    const __bf16* __restrict__ A, int lda,
    const __bf16* __restrict__ W, int ldw,
    float* __restrict__ C, int ldc,
    int M, int N, int K)
{
    __shared__ __bf16 As[2][128][40];
    __shared__ __bf16 Bs[2][128][40];

    const int tid  = threadIdx.x;
    const int lane = tid & 31;
    const int wave = tid >> 5;
    const int wm   = wave & 3;   // 4 row-groups of 32
    const int wn   = wave >> 2;  // 2 col-groups of 64
    const int m0   = blockIdx.y * 128;
    const int n0   = blockIdx.x * 128;

    // two 16B chunks per tile per thread (128 rows x 4 chunks = 512 chunks)
    const int cid0 = tid * 2, cid1 = tid * 2 + 1;
    const int ar0 = cid0 >> 2, ac0 = (cid0 & 3) * 8;
    const int ar1 = cid1 >> 2, ac1 = (cid1 & 3) * 8;
    const int bn0 = (n0 + ar0 < N) ? (n0 + ar0) : (N - 1);  // clamp OOB cols
    const int bn1 = (n0 + ar1 < N) ? (n0 + ar1) : (N - 1);

    v8f acc[2][4];
#pragma unroll
    for (int i = 0; i < 2; ++i)
#pragma unroll
        for (int j = 0; j < 4; ++j) acc[i][j] = zero8();

    auto issue = [&](int k0, int buf) {
        async_copy16(&As[buf][ar0][ac0], A + (long)(m0 + ar0) * lda + k0 + ac0);
        async_copy16(&As[buf][ar1][ac1], A + (long)(m0 + ar1) * lda + k0 + ac1);
        async_copy16(&Bs[buf][ar0][ac0], W + (long)bn0 * ldw + k0 + ac0);
        async_copy16(&Bs[buf][ar1][ac1], W + (long)bn1 * ldw + k0 + ac1);
    };

    const int nk = K >> 5;
    issue(0, 0);
    for (int kt = 0; kt < nk; ++kt) {
        wait_async0();
        __syncthreads();                       // tile kt resident in LDS
        if (kt + 1 < nk) issue((kt + 1) << 5, (kt + 1) & 1);
        const __bf16* as = &As[kt & 1][0][0];
        const __bf16* bs = &Bs[kt & 1][0][0];
#pragma unroll
        for (int mt = 0; mt < 2; ++mt) {
            v16bf a = load_frag_a(as, 40, wm * 32 + mt * 16, 0, lane);
#pragma unroll
            for (int nt = 0; nt < 4; ++nt) {
                v16bf b = load_frag_bn(bs, 40, wn * 64 + nt * 16, 0, lane);
                acc[mt][nt] = wmma_bf16(a, b, acc[mt][nt]);
            }
        }
        __syncthreads();                       // done reading buf before reuse
    }

#pragma unroll
    for (int mt = 0; mt < 2; ++mt)
#pragma unroll
        for (int nt = 0; nt < 4; ++nt)
#pragma unroll
            for (int i = 0; i < 8; ++i) {
                int row = m0 + wm * 32 + mt * 16 + i + ((lane >> 4) << 3);
                int col = n0 + wn * 64 + nt * 16 + (lane & 15);
                if (col < N) C[(long)row * ldc + col] = acc[mt][nt][i];
            }
}

// =====================================================================
// RMSNorm: reads f32 row (stride ld), writes bf16 row (stride ldy).
// =====================================================================
__global__ __launch_bounds__(256) void k_rmsnorm_bf16(
    const float* __restrict__ X, int ld, const float* __restrict__ w, int D,
    __bf16* __restrict__ Y, int ldy)
{
    const int row  = blockIdx.x;
    const float* x = X + (long)row * ld;
    __bf16* y = Y + (long)row * ldy;
    __shared__ float red[8];
    const int tid = threadIdx.x, lane = tid & 31, wave = tid >> 5;

    float ss = 0.f;
    for (int d = tid; d < D; d += 256) { float v = x[d]; ss += v * v; }
#pragma unroll
    for (int off = 16; off > 0; off >>= 1) ss += __shfl_xor(ss, off, 32);
    if (lane == 0) red[wave] = ss;
    __syncthreads();
    if (tid == 0) {
        float t = 0.f;
#pragma unroll
        for (int i = 0; i < 8; ++i) t += red[i];
        red[0] = rsqrtf(t / (float)D + 1e-6f);
    }
    __syncthreads();
    float r = red[0];
    for (int d = tid; d < D; d += 256) y[d] = (__bf16)(w[d] * (x[d] * r));
}

// =====================================================================
// RoPE (interleave-perm) + pack Q/K/V to bf16 (b,h,s,d).
// =====================================================================
__global__ __launch_bounds__(128) void k_rope_pack(
    const float* __restrict__ Q, const float* __restrict__ KV,
    const float* __restrict__ CKV, const int* __restrict__ pos,
    __bf16* __restrict__ Qb, __bf16* __restrict__ Kb, __bf16* __restrict__ Vb,
    int S)
{
    const int row = blockIdx.x;
    const int b = row / S, s = row % S;
    const int tid = threadIdx.x;
    __shared__ float kpe[64];
    __shared__ float cosv[32], sinv[32];

    const float p = (float)pos[row];
    if (tid < 32) {
        float inv_freq = __expf(-logf(10000.0f) * (2.0f * (float)tid) / 64.0f);
        float f = p * inv_freq;
        cosv[tid] = __cosf(f);
        sinv[tid] = __sinf(f);
    }
    __syncthreads();

    const float* kr = CKV + (long)row * 576 + 512;
    if (tid < 64) {
        int j   = tid;
        int pj  = (j < 32) ? 2 * j : 2 * (j - 32) + 1;
        int jr  = (j + 32) & 63;
        int pjr = (jr < 32) ? 2 * jr : 2 * (jr - 32) + 1;
        float sgn = (j < 32) ? -1.f : 1.f;
        kpe[j] = kr[pj] * cosv[j & 31] + sgn * kr[pjr] * sinv[j & 31];
    }
    __syncthreads();

    for (int h = 0; h < 16; ++h) {
        const float* qr  = Q  + (long)row * 3072 + h * 192;
        const float* kvr = KV + (long)row * 4096 + h * 256;
        __bf16* qo = Qb + ((long)(b * 16 + h) * S + s) * 192;
        __bf16* ko = Kb + ((long)(b * 16 + h) * S + s) * 192;
        __bf16* vo = Vb + ((long)(b * 16 + h) * S + s) * 128;
        int d = tid;
        qo[d] = (__bf16)qr[d];
        ko[d] = (__bf16)kvr[d];
        vo[d] = (__bf16)kvr[128 + d];
        if (tid < 64) {
            int j   = tid;
            int pj  = (j < 32) ? 2 * j : 2 * (j - 32) + 1;
            int jr  = (j + 32) & 63;
            int pjr = (jr < 32) ? 2 * jr : 2 * (jr - 32) + 1;
            float sgn = (j < 32) ? -1.f : 1.f;
            float qv = qr[128 + pj] * cosv[j & 31] + sgn * qr[128 + pjr] * sinv[j & 31];
            qo[128 + j] = (__bf16)qv;
            ko[128 + j] = (__bf16)kpe[j];
        }
    }
}

// =====================================================================
// Flash causal MLA attention. 4 waves, 64 q rows, 64-key tiles streamed
// via async global->LDS; QK^T and P·V on v_wmma_f32_16x16x32_bf16.
// Output bf16 (b,s,h*128+d) rows of width 2048.
// =====================================================================
__global__ __launch_bounds__(128) void k_mla_attn(
    const __bf16* __restrict__ Qb, const __bf16* __restrict__ Kb,
    const __bf16* __restrict__ Vb, __bf16* __restrict__ Out, int S)
{
    const int qt = blockIdx.x, h = blockIdx.y, b = blockIdx.z;
    __shared__ __bf16 Qs[64][200];
    __shared__ __bf16 Ks[64][200];
    __shared__ __bf16 Vs[64][136];
    __shared__ float  Ssc[64][68];
    __shared__ __bf16 Pb[64][72];
    __shared__ float  mRow[64], lRow[64], aRow[64];

    const int tid = threadIdx.x, lane = tid & 31, wave = tid >> 5;
    const long headQK = (long)(b * 16 + h) * S * 192;
    const __bf16* Qg = Qb + headQK + (long)qt * 64 * 192;
    const __bf16* Kg = Kb + headQK;
    const __bf16* Vg = Vb + (long)(b * 16 + h) * S * 128;

    // async-stage Q tile: 64x192 bf16 = 1536 16B chunks, 12 per thread
#pragma unroll
    for (int i = 0; i < 12; ++i) {
        int id = tid + i * 128;
        int r = id / 24, c = (id % 24) * 8;
        async_copy16(&Qs[r][c], Qg + r * 192 + c);
    }
    if (tid < 64) { mRow[tid] = -3.0e38f; lRow[tid] = 0.f; }

    v8f o[8];
#pragma unroll
    for (int i = 0; i < 8; ++i) o[i] = zero8();

    const float scale = 0.07216878364870323f;  // 192^-0.5

    for (int kt = 0; kt <= qt; ++kt) {
        // async-stage K tile (12 chunks/thr) and V tile (8 chunks/thr)
#pragma unroll
        for (int i = 0; i < 12; ++i) {
            int id = tid + i * 128;
            int r = id / 24, c = (id % 24) * 8;
            async_copy16(&Ks[r][c], Kg + (long)kt * 64 * 192 + r * 192 + c);
        }
#pragma unroll
        for (int i = 0; i < 8; ++i) {
            int id = tid + i * 128;
            int r = id / 16, c = (id % 16) * 8;
            async_copy16(&Vs[r][c], Vg + (long)kt * 64 * 128 + r * 128 + c);
        }
        wait_async0();
        __syncthreads();

        // scores: each wave 16 q-rows x 64 keys, d=192 (6 K-steps)
        v8f sc[4];
#pragma unroll
        for (int nt = 0; nt < 4; ++nt) sc[nt] = zero8();
#pragma unroll
        for (int ks = 0; ks < 6; ++ks) {
            v16bf a = load_frag_a(&Qs[0][0], 200, wave * 16, ks * 32, lane);
#pragma unroll
            for (int nt = 0; nt < 4; ++nt) {
                v16bf kb = load_frag_bn(&Ks[0][0], 200, nt * 16, ks * 32, lane);
                sc[nt] = wmma_bf16(a, kb, sc[nt]);
            }
        }
#pragma unroll
        for (int nt = 0; nt < 4; ++nt)
#pragma unroll
            for (int i = 0; i < 8; ++i) {
                int r = wave * 16 + i + ((lane >> 4) << 3);
                int c = nt * 16 + (lane & 15);
                int qi = qt * 64 + r, ki = kt * 64 + c;
                float v = sc[nt][i] * scale;
                Ssc[r][c] = (ki <= qi) ? v : -3.4e38f;
            }
        __syncthreads();

        // online softmax per row
        if (tid < 64) {
            int r = tid;
            float m_old = mRow[r], mx = m_old;
            for (int c = 0; c < 64; ++c) mx = fmaxf(mx, Ssc[r][c]);
            float alpha = __expf(m_old - mx);
            float sum = 0.f;
            for (int c = 0; c < 64; ++c) {
                float pv = __expf(Ssc[r][c] - mx);
                sum += pv;
                Pb[r][c] = (__bf16)pv;
            }
            mRow[r] = mx;
            aRow[r] = alpha;
            lRow[r] = lRow[r] * alpha + sum;
        }
        __syncthreads();

        // rescale running output, then O += P @ V
#pragma unroll
        for (int i = 0; i < 8; ++i) {
            int r = wave * 16 + i + ((lane >> 4) << 3);
            float al = aRow[r];
#pragma unroll
            for (int nt = 0; nt < 8; ++nt) o[nt][i] *= al;
        }
#pragma unroll
        for (int ks = 0; ks < 2; ++ks) {
            v16bf a = load_frag_a(&Pb[0][0], 72, wave * 16, ks * 32, lane);
#pragma unroll
            for (int nt = 0; nt < 8; ++nt) {
                v16bf vb = load_frag_bk(&Vs[0][0], 136, nt * 16, ks * 32, lane);
                o[nt] = wmma_bf16(a, vb, o[nt]);
            }
        }
        __syncthreads();
    }

#pragma unroll
    for (int nt = 0; nt < 8; ++nt)
#pragma unroll
        for (int i = 0; i < 8; ++i) {
            int r = wave * 16 + i + ((lane >> 4) << 3);
            int c = nt * 16 + (lane & 15);
            float inv = 1.0f / lRow[r];
            Out[((long)(b * S + qt * 64 + r)) * 2048 + h * 128 + c] =
                (__bf16)(o[nt][i] * inv);
        }
}

// =====================================================================
extern "C" void kernel_launch(void* const* d_in, const int* in_sizes, int n_in,
                              void* d_out, int out_size, void* d_ws, size_t ws_size,
                              hipStream_t stream) {
    (void)in_sizes; (void)n_in; (void)out_size; (void)ws_size;
    const float* h       = (const float*)d_in[0];  // (2,1024,2048)
    const int*   pos     = (const int*)  d_in[1];  // (2,1024)
    const float* q_a_w   = (const float*)d_in[2];  // (1536,2048)
    const float* q_a_ln  = (const float*)d_in[3];  // (1536,)
    const float* q_b_w   = (const float*)d_in[4];  // (3072,1536)
    const float* kv_a_w  = (const float*)d_in[5];  // (576,2048)
    const float* kv_a_ln = (const float*)d_in[6];  // (512,)
    const float* kv_b_w  = (const float*)d_in[7];  // (4096,512)
    const float* o_w     = (const float*)d_in[8];  // (2048,2048)
    float* out = (float*)d_out;

    const int B = 2, S = 1024, M = B * S;  // 2048 rows

    char* ws = (char*)d_ws;
    size_t off = 0;
    auto carve = [&](size_t bytes) -> void* {
        void* p = ws + off;
        off += (bytes + 255) & ~(size_t)255;
        return p;
    };
    __bf16* hb    = (__bf16*)carve((size_t)M * 2048 * 2);
    __bf16* qawb  = (__bf16*)carve((size_t)1536 * 2048 * 2);
    __bf16* qbwb  = (__bf16*)carve((size_t)3072 * 1536 * 2);
    __bf16* kvawb = (__bf16*)carve((size_t)576 * 2048 * 2);
    __bf16* kvbwb = (__bf16*)carve((size_t)4096 * 512 * 2);
    __bf16* owb   = (__bf16*)carve((size_t)2048 * 2048 * 2);
    float*  q_a   = (float*) carve((size_t)M * 1536 * 4);
    float*  ckv   = (float*) carve((size_t)M * 576 * 4);
    __bf16* q_an  = (__bf16*)carve((size_t)M * 1536 * 2);
    __bf16* ckvn  = (__bf16*)carve((size_t)M * 512 * 2);
    float*  q     = (float*) carve((size_t)M * 3072 * 4);
    float*  kv    = (float*) carve((size_t)M * 4096 * 4);
    __bf16* Qb    = (__bf16*)carve((size_t)M * 16 * 192 * 2);
    __bf16* Kb    = (__bf16*)carve((size_t)M * 16 * 192 * 2);
    __bf16* Vb    = (__bf16*)carve((size_t)M * 16 * 128 * 2);
    __bf16* attnb = (__bf16*)carve((size_t)M * 2048 * 2);

    // one-shot f32 -> bf16 converts (L2-resident streams)
    k_f32_to_bf16<<<1024, 256, 0, stream>>>(h,      hb,    (long)M * 2048);
    k_f32_to_bf16<<<1024, 256, 0, stream>>>(q_a_w,  qawb,  (long)1536 * 2048);
    k_f32_to_bf16<<<1024, 256, 0, stream>>>(q_b_w,  qbwb,  (long)3072 * 1536);
    k_f32_to_bf16<<<1024, 256, 0, stream>>>(kv_a_w, kvawb, (long)576 * 2048);
    k_f32_to_bf16<<<1024, 256, 0, stream>>>(kv_b_w, kvbwb, (long)4096 * 512);
    k_f32_to_bf16<<<1024, 256, 0, stream>>>(o_w,    owb,   (long)2048 * 2048);

    // q_a = h @ q_a_w^T ; ckv = h @ kv_a_w^T
    k_gemm_bf16<<<dim3(1536 / 128, M / 128), 256, 0, stream>>>(
        hb, 2048, qawb, 2048, q_a, 1536, M, 1536, 2048);
    k_gemm_bf16<<<dim3((576 + 127) / 128, M / 128), 256, 0, stream>>>(
        hb, 2048, kvawb, 2048, ckv, 576, M, 576, 2048);

    // rmsnorm -> bf16 operands for the next GEMMs
    k_rmsnorm_bf16<<<M, 256, 0, stream>>>(q_a, 1536, q_a_ln, 1536, q_an, 1536);
    k_rmsnorm_bf16<<<M, 256, 0, stream>>>(ckv, 576, kv_a_ln, 512, ckvn, 512);

    // q = q_a_n @ q_b_w^T ; kv = c_n @ kv_b_w^T
    k_gemm_bf16<<<dim3(3072 / 128, M / 128), 256, 0, stream>>>(
        q_an, 1536, qbwb, 1536, q, 3072, M, 3072, 1536);
    k_gemm_bf16<<<dim3(4096 / 128, M / 128), 256, 0, stream>>>(
        ckvn, 512, kvbwb, 512, kv, 4096, M, 4096, 512);

    // RoPE + pack to bf16 (b,h,s,d)
    k_rope_pack<<<M, 128, 0, stream>>>(q, kv, ckv, pos, Qb, Kb, Vb, S);

    // causal attention -> bf16
    k_mla_attn<<<dim3(S / 64, 16, B), 128, 0, stream>>>(Qb, Kb, Vb, attnb, S);

    // out = attn @ o_w^T (f32 result)
    k_gemm_bf16<<<dim3(2048 / 128, M / 128), 256, 0, stream>>>(
        attnb, 2048, owb, 2048, out, 2048, M, 2048, 2048);
}
